// QuantizedLinear_41223096107495
// MI455X (gfx1250) — 8-bit matrix operand order repaired and checked by reading, compiled, not run on hardware
//
#include <hip/hip_runtime.h>
#include <cstdint>
#include <cstddef>

#define MM 8192
#define KK 4096
#define NN 16384

typedef int v8i __attribute__((ext_vector_type(8)));
typedef int v4i_v __attribute__((vector_size(16)));

#if defined(__has_builtin)
#if __has_builtin(__builtin_amdgcn_global_load_async_to_lds_b128) && \
    __has_builtin(__builtin_amdgcn_s_wait_asynccnt)
#define USE_ASYNC_LDS 1
#endif
#endif

#if defined(USE_ASYNC_LDS)
#define ASYNC_CP_B128(gsrc, ldst)                                        \
  __builtin_amdgcn_global_load_async_to_lds_b128(                        \
      (__attribute__((address_space(1))) v4i_v*)(gsrc),                  \
      (__attribute__((address_space(3))) v4i_v*)(ldst), 0, 0)
#endif

__global__ __launch_bounds__(256) void quant_rows(const float* __restrict__ x,
                                                  int8_t* __restrict__ qx,
                                                  float* __restrict__ sx) {
  const int m = blockIdx.x;
  const int tid = threadIdx.x;
  const float4* xr = reinterpret_cast<const float4*>(x + (size_t)m * KK);

  float4 v[4];
  float amax = 0.0f;
#pragma unroll
  for (int i = 0; i < 4; ++i) {
    v[i] = xr[tid + i * 256];
    amax = fmaxf(amax, fmaxf(fmaxf(fabsf(v[i].x), fabsf(v[i].y)),
                             fmaxf(fabsf(v[i].z), fabsf(v[i].w))));
  }
#pragma unroll
  for (int off = 16; off >= 1; off >>= 1)
    amax = fmaxf(amax, __shfl_xor(amax, off, 32));

  __shared__ float red[8];
  if ((tid & 31) == 0) red[tid >> 5] = amax;
  __syncthreads();
  if (tid == 0) {
    float a = red[0];
#pragma unroll
    for (int w = 1; w < 8; ++w) a = fmaxf(a, red[w]);
    red[0] = a;
    sx[m] = a * (1.0f / 127.0f);
  }
  __syncthreads();
  amax = red[0];
  const float inv = (amax > 0.0f) ? (127.0f / amax) : 0.0f;

  int* qrow = reinterpret_cast<int*>(qx + (size_t)m * KK);
#pragma unroll
  for (int i = 0; i < 4; ++i) {
    int qa = (int)rintf(v[i].x * inv);
    int qb = (int)rintf(v[i].y * inv);
    int qc = (int)rintf(v[i].z * inv);
    int qd = (int)rintf(v[i].w * inv);
    qa = max(-127, min(127, qa));
    qb = max(-127, min(127, qb));
    qc = max(-127, min(127, qc));
    qd = max(-127, min(127, qd));
    qrow[tid + i * 256] = (qa & 255) | ((qb & 255) << 8) |
                          ((qc & 255) << 16) | ((qd & 255) << 24);
  }
}

__global__ __launch_bounds__(256) void pack_w(const int* __restrict__ wq,
                                              int8_t* __restrict__ qw) {
  const size_t idx = (size_t)blockIdx.x * 256 + threadIdx.x;
  const int4 v = reinterpret_cast<const int4*>(wq)[idx];
  reinterpret_cast<int*>(qw)[idx] = (v.x & 255) | ((v.y & 255) << 8) |
                                    ((v.z & 255) << 16) | ((v.w & 255) << 24);
}

#define LDS_PITCH 112
#define NKT (KK / 64)

__global__ __launch_bounds__(256) void gemm_i8(const int8_t* __restrict__ qx,
                                               const int8_t* __restrict__ qw,
                                               const float* __restrict__ sx,
                                               const float* __restrict__ wscale,
                                               const float* __restrict__ bias,
                                               float* __restrict__ out) {
  const int tid = threadIdx.x;
  const int lane = tid & 31;
  const int wave = tid >> 5;
  const int wm = wave & 1;
  const int wn = wave >> 1;
  const int half = lane >> 4;
  const int l16 = lane & 15;

  const int n0 = blockIdx.x * 128;
  const int m0 = blockIdx.y * 128;

  const int r0 = tid >> 2;
  const int k0c = (tid & 3) * 16;
  const int r1 = (tid + 256) >> 2;
  const int k1c = ((tid + 256) & 3) * 16;

  const int8_t* gA0 = qx + (size_t)(m0 + r0) * KK + k0c;
  const int8_t* gA1 = qx + (size_t)(m0 + r1) * KK + k1c;
  const int8_t* gB0 = qw + (size_t)(n0 + r0) * KK + k0c;
  const int8_t* gB1 = qw + (size_t)(n0 + r1) * KK + k1c;

  v8i acc[4][2] = {};

  auto compute_slab = [&](const char* bufA, const char* bufB) {
    v8i a[4];
#pragma unroll
    for (int mt = 0; mt < 4; ++mt) {
      const char* ab = &bufA[(wm * 64 + mt * 16 + l16) * LDS_PITCH + half * 16];
      int4 alo = *reinterpret_cast<const int4*>(ab + 0);
      int4 ahi = *reinterpret_cast<const int4*>(ab + 32);
      v8i av;
      av[0] = alo.x; av[1] = alo.y; av[2] = alo.z; av[3] = alo.w;
      av[4] = ahi.x; av[5] = ahi.y; av[6] = ahi.z; av[7] = ahi.w;
      a[mt] = av;
    }

    v8i b[2];
#pragma unroll
    for (int nt = 0; nt < 2; ++nt) {
      const char* bb = &bufB[(wn * 32 + nt * 16 + l16) * LDS_PITCH + half * 16];
      int4 lo = *reinterpret_cast<const int4*>(bb + 0);
      int4 hi = *reinterpret_cast<const int4*>(bb + 32);
      v8i bv;
      bv[0] = lo.x; bv[1] = lo.y; bv[2] = lo.z; bv[3] = lo.w;
      bv[4] = hi.x; bv[5] = hi.y; bv[6] = hi.z; bv[7] = hi.w;
      b[nt] = bv;
    }

#pragma unroll
    for (int mt = 0; mt < 4; ++mt)
#pragma unroll
      for (int nt = 0; nt < 2; ++nt)
        acc[mt][nt] = __builtin_amdgcn_wmma_i32_16x16x64_iu8(
            true, a[mt], true, b[nt], acc[mt][nt], false, false);
  };

#if defined(USE_ASYNC_LDS)
  __shared__ __align__(16) char lA[2][128 * LDS_PITCH];
  __shared__ __align__(16) char lB[2][128 * LDS_PITCH];

#define ASYNC_SLAB(ktArg)                                              \
  do {                                                                 \
    const int _kt = (ktArg);                                           \
    const int _koff = _kt * 64;                                        \
    const int _buf = _kt & 1;                                          \
    ASYNC_CP_B128(gA0 + _koff, &lA[_buf][r0 * LDS_PITCH + k0c]);       \
    ASYNC_CP_B128(gA1 + _koff, &lA[_buf][r1 * LDS_PITCH + k1c]);       \
    ASYNC_CP_B128(gB0 + _koff, &lB[_buf][r0 * LDS_PITCH + k0c]);       \
    ASYNC_CP_B128(gB1 + _koff, &lB[_buf][r1 * LDS_PITCH + k1c]);       \
  } while (0)

  ASYNC_SLAB(0);

#pragma clang loop unroll(disable)
  for (int kt = 0; kt < NKT - 1; ++kt) {
    ASYNC_SLAB(kt + 1);
    __builtin_amdgcn_s_wait_asynccnt(4);
    __syncthreads();
    const char* bufA = (kt & 1) ? lA[1] : lA[0];
    const char* bufB = (kt & 1) ? lB[1] : lB[0];
    compute_slab(bufA, bufB);
    __syncthreads();
  }

  __builtin_amdgcn_s_wait_asynccnt(0);
  __syncthreads();
  compute_slab(lA[1], lB[1]);

#else
  __shared__ __align__(16) char lAbuf[128 * LDS_PITCH];
  __shared__ __align__(16) char lBbuf[128 * LDS_PITCH];

  int4 pa0 = *reinterpret_cast<const int4*>(gA0);
  int4 pa1 = *reinterpret_cast<const int4*>(gA1);
  int4 pb0 = *reinterpret_cast<const int4*>(gB0);
  int4 pb1 = *reinterpret_cast<const int4*>(gB1);

#pragma clang loop unroll(disable)
  for (int kt = 0; kt < NKT; ++kt) {
    __syncthreads();
    *reinterpret_cast<int4*>(&lAbuf[r0 * LDS_PITCH + k0c]) = pa0;
    *reinterpret_cast<int4*>(&lAbuf[r1 * LDS_PITCH + k1c]) = pa1;
    *reinterpret_cast<int4*>(&lBbuf[r0 * LDS_PITCH + k0c]) = pb0;
    *reinterpret_cast<int4*>(&lBbuf[r1 * LDS_PITCH + k1c]) = pb1;
    __syncthreads();

    if (kt + 1 < NKT) {
      const int koff = (kt + 1) * 64;
      pa0 = *reinterpret_cast<const int4*>(gA0 + koff);
      pa1 = *reinterpret_cast<const int4*>(gA1 + koff);
      pb0 = *reinterpret_cast<const int4*>(gB0 + koff);
      pb1 = *reinterpret_cast<const int4*>(gB1 + koff);
    }
    compute_slab(lAbuf, lBbuf);
  }
#endif

#pragma unroll
  for (int mt = 0; mt < 4; ++mt) {
    float srow[8];
#pragma unroll
    for (int r = 0; r < 8; ++r)
      srow[r] = sx[m0 + wm * 64 + mt * 16 + half * 8 + r];
#pragma unroll
    for (int nt = 0; nt < 2; ++nt) {
      const int gn = n0 + wn * 32 + nt * 16 + l16;
      const float wsc = wscale[gn];
      const float bs = bias[gn];
#pragma unroll
      for (int r = 0; r < 8; ++r) {
        const int gm = m0 + wm * 64 + mt * 16 + half * 8 + r;
        out[(size_t)gm * NN + gn] = (float)acc[mt][nt][r] * srow[r] * wsc + bs;
      }
    }
  }
}

extern "C" void kernel_launch(void* const* d_in, const int* in_sizes, int n_in,
                              void* d_out, int out_size, void* d_ws, size_t ws_size,
                              hipStream_t stream) {
  (void)in_sizes; (void)n_in; (void)out_size; (void)ws_size;

  const float* x      = (const float*)d_in[0];
  const int*   wq     = (const int*)d_in[1];
  const float* wscale = (const float*)d_in[2];
  const float* bias   = (const float*)d_in[3];
  float* out = (float*)d_out;

  int8_t* qx = (int8_t*)d_ws;
  int8_t* qw = qx + (size_t)MM * KK;
  float*  sx = (float*)(qw + (size_t)NN * KK);

  quant_rows<<<MM, 256, 0, stream>>>(x, qx, sx);
  pack_w<<<(NN * (size_t)KK) / (4 * 256), 256, 0, stream>>>(wq, qw);
  dim3 grid(NN / 128, MM / 128);
  gemm_i8<<<grid, 256, 0, stream>>>(qx, qw, sx, wscale, bias, out);
}
